// EfficientMultiHeadAttention_45260365365398
// MI455X (gfx1250) — matrix operand order repaired and checked by reading, not compiled, not run on hardware
//
#include <hip/hip_runtime.h>
#include <hip/hip_bf16.h>
#include <stdint.h>

// ---------------------------------------------------------------------------
// Types for CDNA5 WMMA (wave32): v_wmma_f32_16x16x32_bf16
// ---------------------------------------------------------------------------
typedef __bf16 bf16_t;
typedef bf16_t v16bf __attribute__((ext_vector_type(16)));
typedef bf16_t v8bf  __attribute__((ext_vector_type(8)));
typedef float  v8f   __attribute__((ext_vector_type(8)));

#define BATCH 4
#define CDIM  512
#define HDIM  64
#define WDIM  64
#define HEADS 8
#define DHEAD 64
#define NHC   8      // chunks per row/col (64/8)
#define MTOK  64     // tokens per chunk (8x8)
#define NTOK  16384  // B*H*W

__device__ __forceinline__ v16bf combine16(v8bf lo, v8bf hi) {
  v16bf r;
#pragma unroll
  for (int i = 0; i < 8; ++i) { r[i] = lo[i]; r[i + 8] = hi[i]; }
  return r;
}

// A fragment 16x32 bf16, row-major source (ld elements): lane L -> row L&15;
// lanes 0-15 hold K {0..7,16..23}, lanes 16-31 hold K {8..15,24..31}.
__device__ __forceinline__ v16bf load_a(const bf16_t* src, int ld, int k0,
                                        int l16, int half) {
  const bf16_t* p = src + l16 * ld + k0 + (half ? 8 : 0);
  v8bf lo = *(const v8bf*)p;
  v8bf hi = *(const v8bf*)(p + 16);
  return combine16(lo, hi);
}

// B fragment 32x16 bf16, columns stored row-major (src[col*ld + k]):
// lane L -> column L&15; lanes 0-15 hold K {0..7,16..23}, lanes 16-31 hold K {8..15,24..31}.
__device__ __forceinline__ v16bf load_b(const bf16_t* src, int ld, int k0,
                                        int l16, int half) {
  const bf16_t* p = src + l16 * ld + k0 + (half ? 8 : 0);
  v8bf lo = *(const v8bf*)p;
  v8bf hi = *(const v8bf*)(p + 16);
  return combine16(lo, hi);
}

__device__ __forceinline__ v8f wmma_bf16(v16bf a, v16bf b, v8f c) {
  return __builtin_amdgcn_wmma_f32_16x16x32_bf16(false, a, false, b,
                                                 (short)0, c, false, false);
}

// ---------------------------------------------------------------------------
// Kernel 0a: x [B,C,H,W] f32 -> xb [token=(b,y,x)][C] bf16
// ---------------------------------------------------------------------------
__global__ __launch_bounds__(256) void convert_x(const float* __restrict__ x,
                                                 bf16_t* __restrict__ xb) {
  size_t i = (size_t)blockIdx.x * 256 + threadIdx.x;  // over B*C*H*W
  int xx = (int)(i & 63);
  int yy = (int)((i >> 6) & 63);
  int cc = (int)((i >> 12) & 511);
  int bb = (int)(i >> 21);
  xb[((size_t)((bb * 64 + yy) * 64 + xx)) * 512 + cc] = (bf16_t)x[i];
}

// Kernel 0b: generic f32 -> bf16
__global__ __launch_bounds__(256) void convert_w(const float* __restrict__ w,
                                                 bf16_t* __restrict__ wb, int n) {
  int i = blockIdx.x * 256 + threadIdx.x;
  if (i < n) wb[i] = (bf16_t)w[i];
}

// ---------------------------------------------------------------------------
// Kernel 1: QKV GEMM  qkv = xb @ w_qkv^T + b_qkv
// One block per chunk (b,cy,cx). 8 waves: rs=wave&3 (16-row strip of the 64
// chunk tokens, gathered from token-major xb), ch=wave>>2 (half of 96 N-tiles).
// Epilogue scatters into per-(head,chunk) q[tok][d], k[tok][d], vT[d][tok].
// ---------------------------------------------------------------------------
__global__ __launch_bounds__(256) void qkv_gemm(const bf16_t* __restrict__ xb,
                                                const bf16_t* __restrict__ wqb,
                                                const float* __restrict__ bqkv,
                                                bf16_t* __restrict__ qbuf,
                                                bf16_t* __restrict__ kbuf,
                                                bf16_t* __restrict__ vbuf) {
  int blk  = blockIdx.x;            // b*64 + cidx
  int b    = blk >> 6;
  int cidx = blk & 63;
  int cy = cidx >> 3, cx = cidx & 7;
  int wave = threadIdx.x >> 5;
  int lane = threadIdx.x & 31;
  int half = lane >> 4, l16 = lane & 15;
  int rs = wave & 3;                // row strip (16 tokens)
  int ch = wave >> 2;               // column half

  int t  = rs * 16 + l16;           // this lane's A-row token within chunk
  int yy = cy * 8 + (t >> 3);
  int xx = cx * 8 + (t & 7);
  const bf16_t* arow = xb + ((size_t)((b * 64 + yy) * 64 + xx)) * 512;

  for (int g = 0; g < 6; ++g) {
    int ct0 = ch * 48 + g * 8;
    v8f acc[8] = {};
#pragma unroll 4
    for (int ks = 0; ks < 512; ks += 32) {
      const bf16_t* pa = arow + ks + (half ? 8 : 0);
      v16bf a = combine16(*(const v8bf*)pa, *(const v8bf*)(pa + 16));
#pragma unroll
      for (int u = 0; u < 8; ++u) {
        v16bf bb = load_b(wqb + (size_t)(ct0 + u) * 16 * 512, 512, ks, l16, half);
        acc[u] = wmma_bf16(a, bb, acc[u]);
      }
    }
#pragma unroll
    for (int u = 0; u < 8; ++u) {
      int obase = (ct0 + u) * 16;           // 16-col tile never straddles q/k/v or head
      int which = obase >> 9;               // 0=q 1=k 2=v
      int head  = (obase >> 6) & 7;
      int o     = obase + l16;
      int dd    = o & 63;
      float bias = bqkv[o];
      size_t cb = (((size_t)(b * 8 + head)) * 64 + cidx) * 64;  // chunk base (rows of 64)
#pragma unroll
      for (int j = 0; j < 8; ++j) {
        int row = rs * 16 + j + half * 8;   // token within chunk
        bf16_t bv = (bf16_t)(acc[u][j] + bias);
        if (which == 0)       qbuf[(cb + row) * 64 + dd] = bv;
        else if (which == 1)  kbuf[(cb + row) * 64 + dd] = bv;
        else                  vbuf[(cb + dd) * 64 + row] = bv;   // transposed
      }
    }
  }
}

// ---------------------------------------------------------------------------
// Kernel 2: sliding-chunk attention (flash style, online softmax).
// One block per (b,head,cy,cx); 4 waves, each owns a 16-query strip over the
// 3x3 neighbor chunks (576 keys). Invalid border neighbors are skipped
// (== -1e9 mask, since exp underflows to 0 against the always-valid center).
// ---------------------------------------------------------------------------
__global__ __launch_bounds__(128) void attn_kernel(const bf16_t* __restrict__ qbuf,
                                                   const bf16_t* __restrict__ kbuf,
                                                   const bf16_t* __restrict__ vbuf,
                                                   bf16_t* __restrict__ aout) {
  __shared__ bf16_t plds[4 * 16 * 64];      // per-wave P strip [16 q][64 keys]
  int blk  = blockIdx.x;                    // (b*8+head)*64 + cidx
  int cidx = blk & 63;
  int bh   = blk >> 6;
  int cy = cidx >> 3, cx = cidx & 7;
  int b = bh >> 3, head = bh & 7;
  int wave = threadIdx.x >> 5;
  int lane = threadIdx.x & 31;
  int half = lane >> 4, l16 = lane & 15;
  const float scale = 0.125f;               // 1/sqrt(64)

  const bf16_t* qc = qbuf + (size_t)blk * MTOK * DHEAD + wave * 16 * DHEAD;
  bf16_t* pl = plds + wave * 16 * 64;

  v8f acc[4] = {};
  float rm[8], ls[8];
#pragma unroll
  for (int j = 0; j < 8; ++j) { rm[j] = -1e30f; ls[j] = 0.0f; }

  for (int ni = 0; ni < 3; ++ni) {
    for (int nj = 0; nj < 3; ++nj) {
      int ocy = cy + ni - 1, ocx = cx + nj - 1;
      if (ocy < 0 || ocy >= NHC || ocx < 0 || ocx >= NHC) continue;
      size_t oc = (size_t)bh * 64 + (ocy * 8 + ocx);
      const bf16_t* kc = kbuf + oc * MTOK * DHEAD;   // [tok][d]
      const bf16_t* vc = vbuf + oc * DHEAD * MTOK;   // [d][tok]

      // S = Q * Kn^T   (16 x 64 per wave)
      v8f S[4] = {};
#pragma unroll
      for (int kk = 0; kk < 64; kk += 32) {
        v16bf a = load_a(qc, DHEAD, kk, l16, half);
#pragma unroll
        for (int tt = 0; tt < 4; ++tt) {
          v16bf bk = load_b(kc + (size_t)tt * 16 * DHEAD, DHEAD, kk, l16, half);
          S[tt] = wmma_bf16(a, bk, S[tt]);
        }
      }
      // online softmax update (rows of the C tiles = VGPR index per lane half)
      float m[8];
#pragma unroll
      for (int j = 0; j < 8; ++j) {
        float v = fmaxf(fmaxf(S[0][j], S[1][j]), fmaxf(S[2][j], S[3][j])) * scale;
#pragma unroll
        for (int off = 1; off < 16; off <<= 1)
          v = fmaxf(v, __shfl_xor(v, off, 32));
        m[j] = v;
      }
#pragma unroll
      for (int j = 0; j < 8; ++j) {
        float nm    = fmaxf(rm[j], m[j]);
        float alpha = __expf(rm[j] - nm);
        rm[j] = nm;
        ls[j] *= alpha;
#pragma unroll
        for (int ot = 0; ot < 4; ++ot) acc[ot][j] *= alpha;
        int row = j + half * 8;
#pragma unroll
        for (int tt = 0; tt < 4; ++tt) {
          float pv = __expf(S[tt][j] * scale - nm);
          ls[j] += pv;
          pl[row * 64 + tt * 16 + l16] = (bf16_t)pv;   // stage P for A-fragments
        }
      }
      // acc += P @ V   (same-wave LDS producer/consumer; compiler orders ds ops)
#pragma unroll
      for (int kk = 0; kk < 64; kk += 32) {
        v16bf a = load_a(pl, 64, kk, l16, half);
#pragma unroll
        for (int ot = 0; ot < 4; ++ot) {
          v16bf bv = load_b(vc + (size_t)ot * 16 * MTOK, MTOK, kk, l16, half);
          acc[ot] = wmma_bf16(a, bv, acc[ot]);
        }
      }
    }
  }

  // finalize: reduce row sums across the 16-lane groups, normalize, scatter out
#pragma unroll
  for (int j = 0; j < 8; ++j) {
    float s = ls[j];
#pragma unroll
    for (int off = 1; off < 16; off <<= 1) s += __shfl_xor(s, off, 32);
    float rinv = 1.0f / s;
    int q  = wave * 16 + j + half * 8;
    int yy = cy * 8 + (q >> 3);
    int xx = cx * 8 + (q & 7);
    size_t base = ((size_t)((b * 64 + yy) * 64 + xx)) * 512 + head * 64;
#pragma unroll
    for (int ot = 0; ot < 4; ++ot)
      aout[base + ot * 16 + l16] = (bf16_t)(acc[ot][j] * rinv);
  }
}

// ---------------------------------------------------------------------------
// Kernel 3: projection  out[B,C,H,W] = aout @ w_proj^T + b_proj (transposed store)
// ---------------------------------------------------------------------------
__global__ __launch_bounds__(256) void proj_gemm(const bf16_t* __restrict__ ab,
                                                 const bf16_t* __restrict__ wpb,
                                                 const float* __restrict__ bproj,
                                                 float* __restrict__ out) {
  int wave = threadIdx.x >> 5;
  int lane = threadIdx.x & 31;
  int half = lane >> 4, l16 = lane & 15;
  int mt = blockIdx.x * 8 + wave;            // 1024 M-tiles
  size_t g0 = (size_t)mt * 16;
  const bf16_t* arow = ab + (g0 + l16) * 512;

  for (int g = 0; g < 4; ++g) {
    v8f acc[8] = {};
#pragma unroll 4
    for (int ks = 0; ks < 512; ks += 32) {
      const bf16_t* pa = arow + ks + (half ? 8 : 0);
      v16bf a = combine16(*(const v8bf*)pa, *(const v8bf*)(pa + 16));
#pragma unroll
      for (int u = 0; u < 8; ++u) {
        v16bf bb = load_b(wpb + (size_t)(g * 8 + u) * 16 * 512, 512, ks, l16, half);
        acc[u] = wmma_bf16(a, bb, acc[u]);
      }
    }
#pragma unroll
    for (int u = 0; u < 8; ++u) {
      int c = (g * 8 + u) * 16 + l16;
      float bias = bproj[c];
#pragma unroll
      for (int j = 0; j < 8; ++j) {
        size_t tok = g0 + j + half * 8;
        size_t bb  = tok >> 12;              // batch
        size_t yx  = tok & 4095;             // y*64+x
        out[(bb * 512 + c) * 4096 + yx] = acc[u][j] + bias;
      }
    }
  }
}

// ---------------------------------------------------------------------------
extern "C" void kernel_launch(void* const* d_in, const int* in_sizes, int n_in,
                              void* d_out, int out_size, void* d_ws, size_t ws_size,
                              hipStream_t stream) {
  const float* x     = (const float*)d_in[0];
  const float* wqkv  = (const float*)d_in[1];
  const float* bqkv  = (const float*)d_in[2];
  const float* wproj = (const float*)d_in[3];
  const float* bproj = (const float*)d_in[4];
  float* out = (float*)d_out;

  char* p = (char*)d_ws;
  bf16_t* xb  = (bf16_t*)p; p += (size_t)NTOK * 512 * 2;        // 16.78 MB
  bf16_t* wqb = (bf16_t*)p; p += (size_t)1536 * 512 * 2;        //  1.57 MB
  bf16_t* wpb = (bf16_t*)p; p += (size_t)512 * 512 * 2;         //  0.52 MB
  bf16_t* qb  = (bf16_t*)p; p += (size_t)2048 * 64 * 64 * 2;    // 16.78 MB
  bf16_t* kb  = (bf16_t*)p; p += (size_t)2048 * 64 * 64 * 2;    // 16.78 MB
  bf16_t* vb  = (bf16_t*)p; p += (size_t)2048 * 64 * 64 * 2;    // 16.78 MB
  bf16_t* ab  = (bf16_t*)p; p += (size_t)NTOK * 512 * 2;        // 16.78 MB

  convert_x<<<dim3(NTOK * 512 / 256), dim3(256), 0, stream>>>(x, xb);
  convert_w<<<dim3((1536 * 512 + 255) / 256), dim3(256), 0, stream>>>(wqkv, wqb, 1536 * 512);
  convert_w<<<dim3((512 * 512 + 255) / 256), dim3(256), 0, stream>>>(wproj, wpb, 512 * 512);
  qkv_gemm<<<dim3(BATCH * 64), dim3(256), 0, stream>>>(xb, wqb, bqkv, qb, kb, vb);
  attn_kernel<<<dim3(BATCH * HEADS * 64), dim3(128), 0, stream>>>(qb, kb, vb, ab);
  proj_gemm<<<dim3(128), dim3(256), 0, stream>>>(ab, wpb, bproj, out);
}
